// SubLSTM_8478265442724
// MI455X (gfx1250) — matrix operand order repaired and checked by reading, not compiled, not run on hardware
//
#include <hip/hip_runtime.h>

// ---------------------------------------------------------------------------
// subLSTM scan for MI455X (gfx1250): persistent single-WGP kernel, bf16 WMMA,
// async global->LDS double-buffered activation staging.
// T=2048, B=64, I=256, H=256, 4H=1024, K = I+H = 512.
// ---------------------------------------------------------------------------

typedef __bf16 v16bf __attribute__((ext_vector_type(16)));
typedef __bf16 v8bf  __attribute__((ext_vector_type(8)));
typedef float  v8f   __attribute__((ext_vector_type(8)));

constexpr int T_STEPS = 2048;
constexpr int B_SZ    = 64;
constexpr int I_SZ    = 256;
constexpr int H_SZ    = 256;
constexpr int G_SZ    = 4 * H_SZ;    // 1024 gate columns
constexpr int K_TOT   = I_SZ + H_SZ; // 512
constexpr int AST     = 520;         // LDS A row stride (bf16): 512 + 8 pad
constexpr int AELEM   = B_SZ * AST;  // elems per A buffer (66,560 B)

__device__ __forceinline__ float sigmoidf(float v) {
    return 1.0f / (1.0f + __expf(-v));
}

// Fire one 16B async DMA chunk: global (bf16, contiguous) -> LDS.
// VDST carries the wave-relative LDS byte offset (= low 32 bits of the
// generic pointer to a __shared__ object per the LDS aperture mapping).
__device__ __forceinline__ void async_copy16(const __bf16* g, __bf16* l) {
    unsigned loff = (unsigned)(unsigned long long)(uintptr_t)l;
    unsigned long long ga = (unsigned long long)(uintptr_t)g;
    asm volatile("global_load_async_to_lds_b128 %0, %1, off"
                 :: "v"(loff), "v"(ga) : "memory");
}
__device__ __forceinline__ void wait_async0() {
    asm volatile("s_wait_asynccnt 0x0" ::: "memory");
}

// ---------------------------------------------------------------------------
// Prep A: bf16 weights in exact WMMA B-fragment order.
// Fragment (n_tile, k_iter) = 512 bf16 (1 KB): lane L's 16 contiguous K-values
// at elems [((n_tile*16 + k_iter)*32 + L)*16 .. +15].
// B layout used here (16-bit, 32x16, contiguous K): lanes 0-15 hold K=k0+0..15 of col N=lane;
// lanes 16-31 hold K=k0+16..31 of col N=lane-16.
// Column g of the concatenated weight: k<256 -> w_in[g][k], else w_rec[g][k-256].
// ---------------------------------------------------------------------------
__global__ void sublstm_prep_w(const float* __restrict__ w_in,
                               const float* __restrict__ w_rec,
                               __bf16* __restrict__ Wf) {
    const int total = (G_SZ / 16) * (K_TOT / 32) * 32 * 16;  // 524288
    for (int idx = blockIdx.x * blockDim.x + threadIdx.x; idx < total;
         idx += gridDim.x * blockDim.x) {
        const int e      = idx & 15;
        const int lane   = (idx >> 4) & 31;
        const int k_it   = (idx >> 9) & 15;
        const int n_tile = idx >> 13;
        const int n = n_tile * 16 + (lane & 15);
        const int k = k_it * 32 + ((lane >> 4) << 4) + e;
        const float v = (k < I_SZ) ? w_in[n * I_SZ + k]
                                   : w_rec[n * H_SZ + (k - I_SZ)];
        Wf[idx] = (__bf16)v;
    }
}

// Prep B: whole x tensor f32 -> bf16 (one streaming pass, ~9us at 23.3 TB/s),
// so the per-step activation copy is a raw byte move the async DMA can do.
__global__ void sublstm_prep_x(const float* __restrict__ x,
                               __bf16* __restrict__ xb) {
    const int total = T_STEPS * B_SZ * I_SZ;  // 33,554,432
    for (int i = blockIdx.x * blockDim.x + threadIdx.x; i < total;
         i += gridDim.x * blockDim.x)
        xb[i] = (__bf16)x[i];
}

// ---------------------------------------------------------------------------
// Shared per-wave geometry:
// Wave w: hidden sub-tile j = w%16, M-tiles {2*(w/16), 2*(w/16)+1};
// gate N-tiles {j, j+16, j+32, j+48} so i/o/z/f for its (b,j) patch are
// wave-local -> c lives in VGPRs, h written straight back into the A matrix.
// ---------------------------------------------------------------------------

// === Async variant: double-buffered A, one barrier per step ================
__global__ void __launch_bounds__(1024, 1)
sublstm_scan_async(const __bf16* __restrict__ xb,
                   const __bf16* __restrict__ Wf,
                   const float* __restrict__ bias,
                   float* __restrict__ out) {
    __shared__ __bf16 A[2 * AELEM];   // 133,120 B of 320 KB WGP LDS

    const int tid    = threadIdx.x;
    const int lane   = tid & 31;
    const int wave   = tid >> 5;
    const int j_tile = wave & 15;
    const int m0     = (wave >> 4) * 2;
    const int lhalf  = lane >> 4;
    const int ln     = lane & 15;
    const int jcol   = j_tile * 16 + ln;

    // h_0 = 0 in buffer 0; async-stage x_0 into buffer 0.
    for (int idx = tid; idx < B_SZ * H_SZ; idx += 1024)
        A[(idx >> 8) * AST + I_SZ + (idx & 255)] = (__bf16)0.0f;
#pragma unroll
    for (int c = 0; c < 2; ++c) {
        const int chunk = tid + (c << 10);          // 0..2047 (16B chunks)
        const int row   = chunk >> 5;
        const int col8  = (chunk & 31) << 3;
        async_copy16(xb + row * I_SZ + col8, &A[row * AST + col8]);
    }
    wait_async0();

    const float bi  = bias[jcol];
    const float bo  = bias[H_SZ + jcol];
    const float bz  = bias[2 * H_SZ + jcol];
    const float bf_ = bias[3 * H_SZ + jcol];

    float c0[8], c1[8];
#pragma unroll
    for (int v = 0; v < 8; ++v) { c0[v] = 0.0f; c1[v] = 0.0f; }

    const int row0 = m0 * 16 + ln;          // A-frag rows: M = lane&15
    const int row1 = row0 + 16;

    const __bf16* wbase = Wf + (size_t)(j_tile * 16) * 512 + (size_t)lane * 16;
    constexpr size_t GATE_STRIDE = (size_t)16 * 16 * 512;   // 131072 elems

    __syncthreads();   // A[buf0] = [x_0 | h_0] complete

    for (int t = 0; t < T_STEPS; ++t) {
        const int cur = t & 1;
        const int nxt = cur ^ 1;
        const int ab  = cur * AELEM;
        const int nb  = nxt * AELEM;

        // Kick the async DMA for x_{t+1} into the other buffer; it overlaps
        // the whole 128-WMMA GEMM below (tracked by ASYNCcnt).
        if (t + 1 < T_STEPS) {
            const __bf16* xsrc = xb + (size_t)(t + 1) * (B_SZ * I_SZ);
#pragma unroll
            for (int c = 0; c < 2; ++c) {
                const int chunk = tid + (c << 10);
                const int row   = chunk >> 5;
                const int col8  = (chunk & 31) << 3;
                async_copy16(xsrc + row * I_SZ + col8, &A[nb + row * AST + col8]);
            }
        }

        v8f a00 = {bi, bi, bi, bi, bi, bi, bi, bi};  v8f a10 = a00;    // i
        v8f a01 = {bo, bo, bo, bo, bo, bo, bo, bo};  v8f a11 = a01;    // o
        v8f a02 = {bz, bz, bz, bz, bz, bz, bz, bz};  v8f a12 = a02;    // z
        v8f a03 = {bf_, bf_, bf_, bf_, bf_, bf_, bf_, bf_}; v8f a13 = a03; // f

#pragma unroll 4
        for (int k = 0; k < 16; ++k) {
            const int colb = k * 32 + lhalf * 16;
            const v8bf x0l = *(const v8bf*)&A[ab + row0 * AST + colb];
            const v8bf x0h = *(const v8bf*)&A[ab + row0 * AST + colb + 8];
            const v8bf x1l = *(const v8bf*)&A[ab + row1 * AST + colb];
            const v8bf x1h = *(const v8bf*)&A[ab + row1 * AST + colb + 8];
            const v16bf fa0 = __builtin_shufflevector(x0l, x0h,
                0, 1, 2, 3, 4, 5, 6, 7, 8, 9, 10, 11, 12, 13, 14, 15);
            const v16bf fa1 = __builtin_shufflevector(x1l, x1h,
                0, 1, 2, 3, 4, 5, 6, 7, 8, 9, 10, 11, 12, 13, 14, 15);

            const __bf16* wk = wbase + (size_t)k * 512;
            const v16bf Bi = *(const v16bf*)(wk);
            const v16bf Bo = *(const v16bf*)(wk + GATE_STRIDE);
            const v16bf Bz = *(const v16bf*)(wk + 2 * GATE_STRIDE);
            const v16bf Bf = *(const v16bf*)(wk + 3 * GATE_STRIDE);

            a00 = __builtin_amdgcn_wmma_f32_16x16x32_bf16(false, fa0, false, Bi, (short)0, a00, false, false);
            a10 = __builtin_amdgcn_wmma_f32_16x16x32_bf16(false, fa1, false, Bi, (short)0, a10, false, false);
            a01 = __builtin_amdgcn_wmma_f32_16x16x32_bf16(false, fa0, false, Bo, (short)0, a01, false, false);
            a11 = __builtin_amdgcn_wmma_f32_16x16x32_bf16(false, fa1, false, Bo, (short)0, a11, false, false);
            a02 = __builtin_amdgcn_wmma_f32_16x16x32_bf16(false, fa0, false, Bz, (short)0, a02, false, false);
            a12 = __builtin_amdgcn_wmma_f32_16x16x32_bf16(false, fa1, false, Bz, (short)0, a12, false, false);
            a03 = __builtin_amdgcn_wmma_f32_16x16x32_bf16(false, fa0, false, Bf, (short)0, a03, false, false);
            a13 = __builtin_amdgcn_wmma_f32_16x16x32_bf16(false, fa1, false, Bf, (short)0, a13, false, false);
        }

        // subLSTM update; h_{t+1} goes into the *next* buffer (no conflict
        // with other waves still reading buf[cur]).
        // D-tile: VGPR v, lanes 0-15 -> M=v, lanes 16-31 -> M=v+8, N=lane&15.
        float* op = out + (size_t)t * (B_SZ * H_SZ);
#pragma unroll
        for (int v = 0; v < 8; ++v) {
            const int b0 = m0 * 16 + v + lhalf * 8;
            const float cn = sigmoidf(a03[v]) * c0[v] + sigmoidf(a02[v]) - sigmoidf(a00[v]);
            c0[v] = cn;
            const float hn = sigmoidf(cn) - sigmoidf(a01[v]);
            A[nb + b0 * AST + I_SZ + jcol] = (__bf16)hn;
            op[b0 * H_SZ + jcol] = hn;
        }
#pragma unroll
        for (int v = 0; v < 8; ++v) {
            const int b1 = (m0 + 1) * 16 + v + lhalf * 8;
            const float cn = sigmoidf(a13[v]) * c1[v] + sigmoidf(a12[v]) - sigmoidf(a10[v]);
            c1[v] = cn;
            const float hn = sigmoidf(cn) - sigmoidf(a11[v]);
            A[nb + b1 * AST + I_SZ + jcol] = (__bf16)hn;
            op[b1 * H_SZ + jcol] = hn;
        }

        wait_async0();     // x_{t+1} DMA landed (long since, in steady state)
        __syncthreads();   // publish [x_{t+1} | h_{t+1}]; one barrier/step
    }
}

// === Fallback (small workspace): round-1 synchronous staging kernel ========
__global__ void __launch_bounds__(1024, 1)
sublstm_scan_sync(const float* __restrict__ x,
                  const __bf16* __restrict__ Wf,
                  const float* __restrict__ bias,
                  float* __restrict__ out) {
    __shared__ __bf16 A[AELEM];

    const int tid    = threadIdx.x;
    const int lane   = tid & 31;
    const int wave   = tid >> 5;
    const int j_tile = wave & 15;
    const int m0     = (wave >> 4) * 2;
    const int lhalf  = lane >> 4;
    const int ln     = lane & 15;
    const int jcol   = j_tile * 16 + ln;

    for (int idx = tid; idx < B_SZ * H_SZ; idx += 1024)
        A[(idx >> 8) * AST + I_SZ + (idx & 255)] = (__bf16)0.0f;

    const float bi  = bias[jcol];
    const float bo  = bias[H_SZ + jcol];
    const float bz  = bias[2 * H_SZ + jcol];
    const float bf_ = bias[3 * H_SZ + jcol];

    float c0[8], c1[8];
#pragma unroll
    for (int v = 0; v < 8; ++v) { c0[v] = 0.0f; c1[v] = 0.0f; }

    const int row0 = m0 * 16 + ln;
    const int row1 = row0 + 16;
    const __bf16* wbase = Wf + (size_t)(j_tile * 16) * 512 + (size_t)lane * 16;
    constexpr size_t GATE_STRIDE = (size_t)16 * 16 * 512;

    for (int t = 0; t < T_STEPS; ++t) {
        const float* xt = x + (size_t)t * (B_SZ * I_SZ);
#pragma unroll
        for (int r = 0; r < 16; ++r) {
            const int idx = tid + (r << 10);
            A[(idx >> 8) * AST + (idx & 255)] = (__bf16)xt[idx];
        }
        __syncthreads();

        if (t + 1 < T_STEPS)
            __builtin_prefetch(x + (size_t)(t + 1) * (B_SZ * I_SZ) + tid * 16, 0, 0);

        v8f a00 = {bi, bi, bi, bi, bi, bi, bi, bi};  v8f a10 = a00;
        v8f a01 = {bo, bo, bo, bo, bo, bo, bo, bo};  v8f a11 = a01;
        v8f a02 = {bz, bz, bz, bz, bz, bz, bz, bz};  v8f a12 = a02;
        v8f a03 = {bf_, bf_, bf_, bf_, bf_, bf_, bf_, bf_}; v8f a13 = a03;

#pragma unroll 4
        for (int k = 0; k < 16; ++k) {
            const int colb = k * 32 + lhalf * 16;
            const v8bf x0l = *(const v8bf*)&A[row0 * AST + colb];
            const v8bf x0h = *(const v8bf*)&A[row0 * AST + colb + 8];
            const v8bf x1l = *(const v8bf*)&A[row1 * AST + colb];
            const v8bf x1h = *(const v8bf*)&A[row1 * AST + colb + 8];
            const v16bf fa0 = __builtin_shufflevector(x0l, x0h,
                0, 1, 2, 3, 4, 5, 6, 7, 8, 9, 10, 11, 12, 13, 14, 15);
            const v16bf fa1 = __builtin_shufflevector(x1l, x1h,
                0, 1, 2, 3, 4, 5, 6, 7, 8, 9, 10, 11, 12, 13, 14, 15);

            const __bf16* wk = wbase + (size_t)k * 512;
            const v16bf Bi = *(const v16bf*)(wk);
            const v16bf Bo = *(const v16bf*)(wk + GATE_STRIDE);
            const v16bf Bz = *(const v16bf*)(wk + 2 * GATE_STRIDE);
            const v16bf Bf = *(const v16bf*)(wk + 3 * GATE_STRIDE);

            a00 = __builtin_amdgcn_wmma_f32_16x16x32_bf16(false, fa0, false, Bi, (short)0, a00, false, false);
            a10 = __builtin_amdgcn_wmma_f32_16x16x32_bf16(false, fa1, false, Bi, (short)0, a10, false, false);
            a01 = __builtin_amdgcn_wmma_f32_16x16x32_bf16(false, fa0, false, Bo, (short)0, a01, false, false);
            a11 = __builtin_amdgcn_wmma_f32_16x16x32_bf16(false, fa1, false, Bo, (short)0, a11, false, false);
            a02 = __builtin_amdgcn_wmma_f32_16x16x32_bf16(false, fa0, false, Bz, (short)0, a02, false, false);
            a12 = __builtin_amdgcn_wmma_f32_16x16x32_bf16(false, fa1, false, Bz, (short)0, a12, false, false);
            a03 = __builtin_amdgcn_wmma_f32_16x16x32_bf16(false, fa0, false, Bf, (short)0, a03, false, false);
            a13 = __builtin_amdgcn_wmma_f32_16x16x32_bf16(false, fa1, false, Bf, (short)0, a13, false, false);
        }

        __syncthreads();

        float* op = out + (size_t)t * (B_SZ * H_SZ);
#pragma unroll
        for (int v = 0; v < 8; ++v) {
            const int b0 = m0 * 16 + v + lhalf * 8;
            const float cn = sigmoidf(a03[v]) * c0[v] + sigmoidf(a02[v]) - sigmoidf(a00[v]);
            c0[v] = cn;
            const float hn = sigmoidf(cn) - sigmoidf(a01[v]);
            A[b0 * AST + I_SZ + jcol] = (__bf16)hn;
            op[b0 * H_SZ + jcol] = hn;
        }
#pragma unroll
        for (int v = 0; v < 8; ++v) {
            const int b1 = (m0 + 1) * 16 + v + lhalf * 8;
            const float cn = sigmoidf(a13[v]) * c1[v] + sigmoidf(a12[v]) - sigmoidf(a10[v]);
            c1[v] = cn;
            const float hn = sigmoidf(cn) - sigmoidf(a11[v]);
            A[b1 * AST + I_SZ + jcol] = (__bf16)hn;
            op[b1 * H_SZ + jcol] = hn;
        }
    }
}

// ---------------------------------------------------------------------------
// inputs (setup_inputs order): x [T,B,I] f32, w_in [4H,I] f32, b [4H] f32,
// w_rec [4H,H] f32. out: [T,B,H] f32.
// ws: [0,1MB) fragment-ordered bf16 weights; [1MB,1MB+64MB) bf16 x (async path).
// ---------------------------------------------------------------------------
extern "C" void kernel_launch(void* const* d_in, const int* in_sizes, int n_in,
                              void* d_out, int out_size, void* d_ws, size_t ws_size,
                              hipStream_t stream) {
    (void)in_sizes; (void)n_in; (void)out_size;
    const float* x     = (const float*)d_in[0];
    const float* w_in  = (const float*)d_in[1];
    const float* bias  = (const float*)d_in[2];
    const float* w_rec = (const float*)d_in[3];
    float*  out = (float*)d_out;

    __bf16* Wf = (__bf16*)d_ws;
    const size_t WF_BYTES = (size_t)(G_SZ / 16) * (K_TOT / 32) * 512 * 2;   // 1 MB
    const size_t XB_BYTES = (size_t)T_STEPS * B_SZ * I_SZ * 2;              // 64 MB

    sublstm_prep_w<<<256, 256, 0, stream>>>(w_in, w_rec, Wf);

    if (ws_size >= WF_BYTES + XB_BYTES) {
        __bf16* xb = (__bf16*)((char*)d_ws + WF_BYTES);
        sublstm_prep_x<<<2048, 256, 0, stream>>>(x, xb);
        sublstm_scan_async<<<1, 1024, 0, stream>>>(xb, Wf, bias, out);
    } else {
        sublstm_scan_sync<<<1, 1024, 0, stream>>>(x, Wf, bias, out);
    }
}
